// Attention_32959579029639
// MI455X (gfx1250) — matrix operand order repaired and checked by reading, not compiled, not run on hardware
//
#include <hip/hip_runtime.h>
#include <hip/hip_bf16.h>
#include <math.h>

// ---------------------------------------------------------------------------
// PoPE attention for MI455X (gfx1250), bf16 WMMA path with async-LDS staging
// and software-pipelined WMMA loops.
// B=1, S=2048, DIM=1024, HEADS=16, HEAD_DIM=64, EMBED=128.
// ---------------------------------------------------------------------------

typedef __bf16 bf16;
typedef __attribute__((ext_vector_type(16))) __bf16 v16bf;
typedef __attribute__((ext_vector_type(8)))  __bf16 v8bf;
typedef __attribute__((ext_vector_type(8)))  float  v8f;

constexpr int S_LEN = 2048;
constexpr int DIMC  = 1024;
constexpr int HDIM  = 64;
constexpr int EDIM  = 128;   // 2*HDIM
constexpr int CHUNK = 512;   // keys per online-softmax chunk
constexpr int NCHUNK = S_LEN / CHUNK;
constexpr int KSTEP = 64;    // K per staged B tile in projection GEMMs

static __device__ __forceinline__ unsigned short f2bf_bits(float f) {
  union { float f; unsigned u; } v; v.f = f;
  unsigned r = v.u + 0x7fffu + ((v.u >> 16) & 1u);   // RNE
  return (unsigned short)(r >> 16);
}
static __device__ __forceinline__ bf16 f2bf(float f) {
  union { unsigned short s; bf16 b; } o; o.s = f2bf_bits(f);
  return o.b;
}
static __device__ __forceinline__ v16bf cat8(v8bf lo, v8bf hi) {
  return __builtin_shufflevector(lo, hi, 0,1,2,3,4,5,6,7,8,9,10,11,12,13,14,15);
}
static __device__ __forceinline__ v8f wmma_bf16(v16bf a, v16bf b, v8f c) {
  return __builtin_amdgcn_wmma_f32_16x16x32_bf16(false, a, false, b, (short)0, c,
                                                 false, false);
}

// CDNA5 async global->LDS copy (16B per lane), tracked by ASYNCcnt.
static __device__ __forceinline__ void async_ld_b128(const bf16* g,
                                                     const bf16* lds_ptr) {
  unsigned loff = (unsigned)(unsigned long long)lds_ptr;  // low 32 = LDS offset
  unsigned long long ga = (unsigned long long)g;
  asm volatile("global_load_async_to_lds_b128 %0, %1, off"
               :: "v"(loff), "v"(ga) : "memory");
}
static __device__ __forceinline__ void wait_async0() {
  asm volatile("s_wait_asynccnt 0x0" ::: "memory");
}

// ---------------------------------------------------------------------------
// fp32 -> bf16 conversion, 4 elems / thread
// ---------------------------------------------------------------------------
__global__ __launch_bounds__(256) void cvt_bf16_kernel(
    const float* __restrict__ src, bf16* __restrict__ dst, int n4) {
  int i = blockIdx.x * blockDim.x + threadIdx.x;
  if (i < n4) {
    float4 f = ((const float4*)src)[i];
    ushort4 u;
    u.x = f2bf_bits(f.x); u.y = f2bf_bits(f.y);
    u.z = f2bf_bits(f.z); u.w = f2bf_bits(f.w);
    ((ushort4*)dst)[i] = u;
  }
}

// ---------------------------------------------------------------------------
// Fused QKV projection + PoPE embedding.
// Block = one (proj, head, 256-row s-block): 8 waves x (32s x 64n) tiles, all
// sharing one 64x64 weight tile staged in LDS by async copies (double-buffer).
// grid = 3*16*8 = 384 blocks x 256 threads.
// ---------------------------------------------------------------------------
__global__ __launch_bounds__(256) void qkv_pope_kernel(
    const bf16* __restrict__ xb,
    const bf16* __restrict__ wqb, const bf16* __restrict__ wkb,
    const bf16* __restrict__ wvb,
    const float* __restrict__ pope_bias,
    bf16* __restrict__ qe, bf16* __restrict__ ke, bf16* __restrict__ vt) {
  __shared__ bf16 bstage[2][64 * KSTEP];   // 2 x 8 KB

  const int tid  = threadIdx.x;
  const int lane = tid & 31;
  const int wave = tid >> 5;
  const int ln   = lane & 15;
  const int hi8  = (lane >> 4) * 8;
  const int hi16 = (lane >> 4) * 16;

  const int bid  = blockIdx.x;
  const int proj = bid / 128;              // 0=q 1=k 2=v
  const int rem  = bid % 128;
  const int h    = rem >> 3;
  const int s0   = (rem & 7) * 256 + wave * 32;

  const bf16* w     = (proj == 0) ? wqb : (proj == 1) ? wkb : wvb;
  const bf16* wtile = w + (size_t)h * HDIM * DIMC;     // rows o = h*64 + row
  const bf16* xrow0 = xb + (size_t)(s0 + ln) * DIMC;
  const bf16* xrow1 = xb + (size_t)(s0 + 16 + ln) * DIMC;

  // this wave's share of staging: 2 async b128 per k-step
  auto stage = [&](int buf, int k0) {
    #pragma unroll
    for (int q = 0; q < 2; ++q) {
      int t    = (wave * 2 + q) * 32 + lane;   // 0..511
      int row  = t >> 3;                       // 0..63
      int ch   = t & 7;                        // 16B chunk in row
      async_ld_b128(wtile + (size_t)row * DIMC + k0 + ch * 8,
                    &bstage[buf][row * KSTEP + ch * 8]);
    }
  };
  auto loadA = [&](const bf16* rowp, int kc) {
    v8bf lo = *(const v8bf*)(rowp + kc + hi16);
    v8bf hi = *(const v8bf*)(rowp + kc + hi16 + 8);
    return cat8(lo, hi);
  };

  v8f acc[2][4];
  #pragma unroll
  for (int m = 0; m < 2; ++m)
    #pragma unroll
    for (int n = 0; n < 4; ++n) acc[m][n] = (v8f)(0.0f);

  v16bf a[2][4];   // [parity][0=m0k0 1=m1k0 2=m0k32 3=m1k32]
  stage(0, 0);
  a[0][0] = loadA(xrow0, 0);  a[0][1] = loadA(xrow1, 0);
  a[0][2] = loadA(xrow0, 32); a[0][3] = loadA(xrow1, 32);
  wait_async0();
  __syncthreads();

  for (int it = 0; it < DIMC / KSTEP; ++it) {
    const int cbuf = it & 1, nbuf = cbuf ^ 1;
    const int kn = (it + 1) * KSTEP;
    if (kn < DIMC) {
      stage(nbuf, kn);
      a[nbuf][0] = loadA(xrow0, kn);      a[nbuf][1] = loadA(xrow1, kn);
      a[nbuf][2] = loadA(xrow0, kn + 32); a[nbuf][3] = loadA(xrow1, kn + 32);
    }
    // hoist ALL B fragments of this k-step, then run the 16-WMMA block, so
    // the scheduler can retire WMMAs against partial s_wait_dscnt counts.
    v16bf bfrag[2][4];
    #pragma unroll
    for (int ksI = 0; ksI < 2; ++ksI)
      #pragma unroll
      for (int n = 0; n < 4; ++n)
        bfrag[ksI][n] = *(const v16bf*)&bstage[cbuf][(n * 16 + ln) * KSTEP +
                                                     ksI * 32 + hi16];
    #pragma unroll
    for (int ksI = 0; ksI < 2; ++ksI)
      #pragma unroll
      for (int n = 0; n < 4; ++n) {
        acc[0][n] = wmma_bf16(a[cbuf][ksI * 2 + 0], bfrag[ksI][n], acc[0][n]);
        acc[1][n] = wmma_bf16(a[cbuf][ksI * 2 + 1], bfrag[ksI][n], acc[1][n]);
      }
    if (kn < DIMC) {
      wait_async0();
      __syncthreads();
    }
  }

  // epilogue: C-frag element (VGPR j) is row M=j+hi8, col N=ln within n-tile
  #pragma unroll
  for (int m = 0; m < 2; ++m) {
    #pragma unroll
    for (int n = 0; n < 4; ++n) {
      #pragma unroll
      for (int j = 0; j < 8; ++j) {
        const int srow = s0 + m * 16 + j + hi8;
        const int d = n * 16 + ln;
        const float val = acc[m][n][j];
        if (proj == 2) {
          vt[(size_t)(h * HDIM + d) * S_LEN + srow] = f2bf(val);
        } else {
          float mag = (val > 20.f) ? val : log1pf(__expf(val));
          float ang = (float)srow * __expf(-(float)d * (9.210340372f / 64.f));
          if (proj == 1) {
            float bs = pope_bias[h * HDIM + d];
            bs = fminf(fmaxf(bs, -6.28318530718f), 0.f);
            ang += bs;
          }
          float sn, cs;
          __sincosf(ang, &sn, &cs);
          unsigned pk = (unsigned)f2bf_bits(mag * cs) |
                        ((unsigned)f2bf_bits(mag * sn) << 16);
          bf16* dst = ((proj == 0) ? qe : ke) +
                      ((size_t)h * S_LEN + srow) * EDIM + 2 * d;
          *(unsigned*)dst = pk;
        }
      }
    }
  }
}

// ---------------------------------------------------------------------------
// Flash-style attention: one block = (head, 16-query tile), 8 waves.
// Online softmax over 4 key chunks of 512; scores f32 in LDS, probs bf16 in
// LDS; scores and P@V WMMA loops are register double-buffered.
// grid = 16*128 = 2048 blocks x 256 threads
// ---------------------------------------------------------------------------
__global__ __launch_bounds__(256) void attn_kernel(
    const bf16* __restrict__ qe, const bf16* __restrict__ ke,
    const bf16* __restrict__ vt, bf16* __restrict__ ao) {
  __shared__ float sc[16 * CHUNK];       // 32 KB (reused as o-reduction buf)
  __shared__ bf16  pb[16 * CHUNK];       // 16 KB
  __shared__ float mrow[16], lrow[16], arow[16];
  __shared__ float red[16 * 16];

  const int tid  = threadIdx.x;
  const int lane = tid & 31;
  const int wave = tid >> 5;
  const int h    = blockIdx.x >> 7;
  const int s0   = (blockIdx.x & 127) * 16;
  const int ln   = lane & 15;
  const int hi8  = (lane >> 4) * 8;
  const int hi16 = (lane >> 4) * 16;

  if (tid < 16) { mrow[tid] = -3.0e38f; lrow[tid] = 0.f; }

  // A-fragments of q_embed (rows s0..s0+15, K = 0..127), reused all chunks
  const bf16* qrow = qe + ((size_t)h * S_LEN + s0 + ln) * EDIM;
  v16bf aq[4];
  #pragma unroll
  for (int e = 0; e < 4; ++e) {
    v8bf lo = *(const v8bf*)(qrow + e * 32 + hi16);
    v8bf hi = *(const v8bf*)(qrow + e * 32 + hi16 + 8);
    aq[e] = cat8(lo, hi);
  }

  v8f o[4];
  #pragma unroll
  for (int n = 0; n < 4; ++n) o[n] = (v8f)(0.0f);

  const bf16* keh = ke + (size_t)h * S_LEN * EDIM;

  __syncthreads();

  for (int c = 0; c < NCHUNK; ++c) {
    // ---- scores: this wave handles keys [c*512 + wave*64, +64) ----
    const int tb = c * CHUNK + wave * 64;
    v16bf bk[2][4];
    #pragma unroll
    for (int e = 0; e < 4; ++e)
      bk[0][e] = *(const v16bf*)(keh + (size_t)(tb + ln) * EDIM + e * 32 + hi16);
    #pragma unroll
    for (int kt = 0; kt < 4; ++kt) {
      if (kt < 3) {
        const int tn = tb + (kt + 1) * 16;
        #pragma unroll
        for (int e = 0; e < 4; ++e)
          bk[(kt + 1) & 1][e] =
              *(const v16bf*)(keh + (size_t)(tn + ln) * EDIM + e * 32 + hi16);
      }
      v8f sacc = (v8f)(0.0f);
      #pragma unroll
      for (int e = 0; e < 4; ++e) sacc = wmma_bf16(aq[e], bk[kt & 1][e], sacc);
      const int col = wave * 64 + kt * 16 + ln;
      #pragma unroll
      for (int j = 0; j < 8; ++j)
        sc[(j + hi8) * CHUNK + col] = sacc[j] * 0.125f;   // / sqrt(64)
    }
    __syncthreads();

    // ---- online softmax bookkeeping: 16 threads per row ----
    const int r = tid >> 4, sub = tid & 15;
    const float* srow = &sc[r * CHUNK + sub * 32];
    float lm = -3.0e38f;
    #pragma unroll 8
    for (int i = 0; i < 32; ++i) lm = fmaxf(lm, srow[i]);
    red[tid] = lm;
    __syncthreads();
    if (sub == 0) {
      float rm = red[r * 16];
      #pragma unroll
      for (int i = 1; i < 16; ++i) rm = fmaxf(rm, red[r * 16 + i]);
      float mo = mrow[r];
      float mn = fmaxf(mo, rm);
      arow[r] = __expf(mo - mn);
      mrow[r] = mn;
    }
    __syncthreads();
    const float mn = mrow[r];
    bf16* prow = &pb[r * CHUNK + sub * 32];
    float ls = 0.f;
    #pragma unroll 8
    for (int i = 0; i < 32; ++i) {
      float p = __expf(srow[i] - mn);
      prow[i] = f2bf(p);
      ls += p;
    }
    red[tid] = ls;
    __syncthreads();
    if (sub == 0) {
      float ts = 0.f;
      #pragma unroll
      for (int i = 0; i < 16; ++i) ts += red[r * 16 + i];
      lrow[r] = lrow[r] * arow[r] + ts;
    }
    __syncthreads();

    // ---- rescale running O, accumulate P @ V for this wave's 64 keys ----
    float al[8];
    #pragma unroll
    for (int j = 0; j < 8; ++j) al[j] = arow[j + hi8];
    #pragma unroll
    for (int n = 0; n < 4; ++n)
      #pragma unroll
      for (int j = 0; j < 8; ++j) o[n][j] *= al[j];

    // hoist both LDS A-fragments (P) and all global B-fragments (V) up front
    const int tv = c * CHUNK + wave * 64 + hi16;
    v16bf ap[2];
    #pragma unroll
    for (int ksI = 0; ksI < 2; ++ksI) {
      const bf16* pr = &pb[ln * CHUNK + wave * 64 + ksI * 32];
      v8bf plo = *(const v8bf*)(pr + hi16);
      v8bf phi = *(const v8bf*)(pr + hi16 + 8);
      ap[ksI] = cat8(plo, phi);
    }
    v16bf bv[2][4];
    #pragma unroll
    for (int ksI = 0; ksI < 2; ++ksI)
      #pragma unroll
      for (int n = 0; n < 4; ++n)
        bv[ksI][n] = *(const v16bf*)(vt +
                     (size_t)(h * HDIM + n * 16 + ln) * S_LEN + tv + ksI * 32);
    #pragma unroll
    for (int ksI = 0; ksI < 2; ++ksI)
      #pragma unroll
      for (int n = 0; n < 4; ++n) o[n] = wmma_bf16(ap[ksI], bv[ksI][n], o[n]);
    __syncthreads();   // pb/sc reused next chunk
  }

  // ---- cross-wave reduction of 16x64 output tile ----
  float* ored = sc;   // reuse 32 KB
  #pragma unroll
  for (int n = 0; n < 4; ++n)
    #pragma unroll
    for (int j = 0; j < 8; ++j)
      ored[(wave * 16 + j + hi8) * 64 + n * 16 + ln] = o[n][j];
  __syncthreads();

  #pragma unroll
  for (int i = 0; i < 4; ++i) {
    const int e = tid + i * 256;          // 16*64 = 1024 elems
    const int r = e >> 6, d = e & 63;
    float s = 0.f;
    #pragma unroll
    for (int w = 0; w < 8; ++w) s += ored[(w * 16 + r) * 64 + d];
    s /= lrow[r];
    ao[(size_t)(s0 + r) * DIMC + h * HDIM + d] = f2bf(s);
  }
}

// ---------------------------------------------------------------------------
// Output projection: out = ao(2048x1024 bf16) @ wo^T, f32 result.
// Block = 256(s) x 64(o): 8 waves x (32s x 64o), weight tile async-staged in
// LDS (double buffer).  grid = 8*16 = 128 blocks x 256 threads.
// ---------------------------------------------------------------------------
__global__ __launch_bounds__(256) void oproj_kernel(
    const bf16* __restrict__ ao, const bf16* __restrict__ wob,
    float* __restrict__ out) {
  __shared__ bf16 bstage[2][64 * KSTEP];   // 2 x 8 KB

  const int tid  = threadIdx.x;
  const int lane = tid & 31;
  const int wave = tid >> 5;
  const int ln   = lane & 15;
  const int hi8  = (lane >> 4) * 8;
  const int hi16 = (lane >> 4) * 16;

  const int s0 = (blockIdx.x >> 4) * 256 + wave * 32;
  const int o0 = (blockIdx.x & 15) * 64;

  const bf16* wtile = wob + (size_t)o0 * DIMC;
  const bf16* arow0 = ao + (size_t)(s0 + ln) * DIMC;
  const bf16* arow1 = ao + (size_t)(s0 + 16 + ln) * DIMC;

  auto stage = [&](int buf, int k0) {
    #pragma unroll
    for (int q = 0; q < 2; ++q) {
      int t   = (wave * 2 + q) * 32 + lane;
      int row = t >> 3;
      int ch  = t & 7;
      async_ld_b128(wtile + (size_t)row * DIMC + k0 + ch * 8,
                    &bstage[buf][row * KSTEP + ch * 8]);
    }
  };
  auto loadA = [&](const bf16* rowp, int kc) {
    v8bf lo = *(const v8bf*)(rowp + kc + hi16);
    v8bf hi = *(const v8bf*)(rowp + kc + hi16 + 8);
    return cat8(lo, hi);
  };

  v8f acc[2][4];
  #pragma unroll
  for (int m = 0; m < 2; ++m)
    #pragma unroll
    for (int n = 0; n < 4; ++n) acc[m][n] = (v8f)(0.0f);

  v16bf a[2][4];
  stage(0, 0);
  a[0][0] = loadA(arow0, 0);  a[0][1] = loadA(arow1, 0);
  a[0][2] = loadA(arow0, 32); a[0][3] = loadA(arow1, 32);
  wait_async0();
  __syncthreads();

  for (int it = 0; it < DIMC / KSTEP; ++it) {
    const int cbuf = it & 1, nbuf = cbuf ^ 1;
    const int kn = (it + 1) * KSTEP;
    if (kn < DIMC) {
      stage(nbuf, kn);
      a[nbuf][0] = loadA(arow0, kn);      a[nbuf][1] = loadA(arow1, kn);
      a[nbuf][2] = loadA(arow0, kn + 32); a[nbuf][3] = loadA(arow1, kn + 32);
    }
    v16bf bfrag[2][4];
    #pragma unroll
    for (int ksI = 0; ksI < 2; ++ksI)
      #pragma unroll
      for (int n = 0; n < 4; ++n)
        bfrag[ksI][n] = *(const v16bf*)&bstage[cbuf][(n * 16 + ln) * KSTEP +
                                                     ksI * 32 + hi16];
    #pragma unroll
    for (int ksI = 0; ksI < 2; ++ksI)
      #pragma unroll
      for (int n = 0; n < 4; ++n) {
        acc[0][n] = wmma_bf16(a[cbuf][ksI * 2 + 0], bfrag[ksI][n], acc[0][n]);
        acc[1][n] = wmma_bf16(a[cbuf][ksI * 2 + 1], bfrag[ksI][n], acc[1][n]);
      }
    if (kn < DIMC) {
      wait_async0();
      __syncthreads();
    }
  }

  #pragma unroll
  for (int m = 0; m < 2; ++m)
    #pragma unroll
    for (int n = 0; n < 4; ++n)
      #pragma unroll
      for (int j = 0; j < 8; ++j)
        out[(size_t)(s0 + m * 16 + j + hi8) * DIMC + o0 + n * 16 + ln] =
            acc[m][n][j];
}

// ---------------------------------------------------------------------------
extern "C" void kernel_launch(void* const* d_in, const int* in_sizes, int n_in,
                              void* d_out, int out_size, void* d_ws,
                              size_t ws_size, hipStream_t stream) {
  const float* x  = (const float*)d_in[0];
  const float* wq = (const float*)d_in[1];
  const float* wk = (const float*)d_in[2];
  const float* wv = (const float*)d_in[3];
  const float* wo = (const float*)d_in[4];
  const float* pb = (const float*)d_in[5];
  float* out = (float*)d_out;

  char* ws = (char*)d_ws;
  const size_t MB = 1u << 20;
  bf16* xb  = (bf16*)(ws + 0 * MB);    // 2048x1024          (4 MiB)
  bf16* wqb = (bf16*)(ws + 4 * MB);    // 1024x1024          (2 MiB)
  bf16* wkb = (bf16*)(ws + 6 * MB);
  bf16* wvb = (bf16*)(ws + 8 * MB);
  bf16* wob = (bf16*)(ws + 10 * MB);
  bf16* qe  = (bf16*)(ws + 12 * MB);   // (h,s,128)          (8 MiB)
  bf16* ke  = (bf16*)(ws + 20 * MB);   // (h,s,128)          (8 MiB)
  bf16* vt  = (bf16*)(ws + 28 * MB);   // (h,d,s)            (4 MiB)
  bf16* ao  = (bf16*)(ws + 32 * MB);   // (s,c)              (4 MiB)

  const int nx4 = (S_LEN * DIMC) / 4;      // 524288
  const int nw4 = (DIMC * DIMC) / 4;       // 262144
  cvt_bf16_kernel<<<nx4 / 256, 256, 0, stream>>>(x, xb, nx4);
  cvt_bf16_kernel<<<nw4 / 256, 256, 0, stream>>>(wq, wqb, nw4);
  cvt_bf16_kernel<<<nw4 / 256, 256, 0, stream>>>(wk, wkb, nw4);
  cvt_bf16_kernel<<<nw4 / 256, 256, 0, stream>>>(wv, wvb, nw4);
  cvt_bf16_kernel<<<nw4 / 256, 256, 0, stream>>>(wo, wob, nw4);

  qkv_pope_kernel<<<384, 256, 0, stream>>>(xb, wqb, wkb, wvb, pb, qe, ke, vt);
  attn_kernel<<<2048, 256, 0, stream>>>(qe, ke, vt, ao);
  oproj_kernel<<<128, 256, 0, stream>>>(ao, wob, out);
}
